// OctodiffModel_11871289606212
// MI455X (gfx1250) — matrix operand order repaired and checked by reading, not compiled, not run on hardware
//
#include <hip/hip_runtime.h>

typedef __bf16 bf16_t;
typedef __bf16 v16bf __attribute__((ext_vector_type(16)));
typedef __bf16 v8bf  __attribute__((ext_vector_type(8)));
typedef float  v8f   __attribute__((ext_vector_type(8)));

namespace cfg {
constexpr int H = 2048, I = 8192, L = 4;
constexpr int NH = 16, NKV = 4, D = 128, GRP = 4;
constexpr int B = 2, S = 2048, T = B * S;
constexpr float EPS = 1e-6f;
}

// ---------------------------------------------------------------------------
// CDNA5 async global->LDS copy (ASYNCcnt path, CDNA5 ISA)
// LDS byte offset = low 32 bits of the generic address (CDNA5 ISA)
// ---------------------------------------------------------------------------
__device__ __forceinline__ void async_copy_b128(void* lds_dst, const void* gsrc) {
  const unsigned lds_off = (unsigned)(uintptr_t)lds_dst;
  asm volatile("global_load_async_to_lds_b128 %0, %1, off"
               :
               : "v"(lds_off), "v"((unsigned long long)(uintptr_t)gsrc)
               : "memory");
}
__device__ __forceinline__ void wait_async() {
  asm volatile("s_wait_asynccnt 0" ::: "memory");
}

// ---------------------------------------------------------------------------
// WMMA helpers (bf16 A/B, fp32 accumulate), wave32; A/B in contiguous k order, C layout per ISA 7.12.2
// ---------------------------------------------------------------------------
__device__ __forceinline__ v8f wmma_bf16(v16bf a, v16bf b, v8f c) {
  // 8 args: (neg_a, A, neg_b, B, c_mod, C, reuse_a, reuse_b)
  return __builtin_amdgcn_wmma_f32_16x16x32_bf16(false, a, false, b, (short)0, c,
                                                 false, false);
}

__device__ __forceinline__ v8f vzero8() {
  v8f z;
#pragma unroll
  for (int e = 0; e < 8; ++e) z[e] = 0.0f;
  return z;
}

// A fragment: 16x32 bf16, row-major LDS tile with stride ld (elements).
// lane: m = row0 + (lane&15); kb = k0 + (lane>>4)*16
// elems 0..15 -> K = kb+0..15
__device__ __forceinline__ v16bf load_frag_a(const bf16_t* base, int ld, int row0,
                                             int k0, int lane) {
  const int m = row0 + (lane & 15);
  const int kb = k0 + ((lane >> 4) << 4);
  const bf16_t* p = base + m * ld + kb;
  v8bf x = *(const v8bf*)(p);
  v8bf y = *(const v8bf*)(p + 8);
  v16bf a;
#pragma unroll
  for (int e = 0; e < 8; ++e) { a[e] = x[e]; a[e + 8] = y[e]; }
  return a;
}

// B fragment (32x16) from LDS tile holding B as [k][n] row-major, stride ld.
// lane: n = n0 + (lane&15); k = k0 + (lane>>4)*16 + e   (strided gather)
__device__ __forceinline__ v16bf load_frag_b_kn(const bf16_t* base, int ld, int n0,
                                                int k0, int lane) {
  const int n = n0 + (lane & 15);
  const int kb = k0 + ((lane >> 4) << 4);
  v16bf b;
#pragma unroll
  for (int e = 0; e < 16; ++e) b[e] = base[(kb + e) * ld + n];
  return b;
}

// B fragment (32x16) where the LDS tile holds B transposed as [n][k] row-major
// (contiguous in k -> two aligned ds_load_b128). Used for Q*K^T and GEMM B.
__device__ __forceinline__ v16bf load_frag_b_nk(const bf16_t* base, int ld, int n0,
                                                int k0, int lane) {
  const int n = n0 + (lane & 15);
  const int kb = k0 + ((lane >> 4) << 4);
  const bf16_t* p = base + n * ld + kb;
  v8bf x = *(const v8bf*)(p);
  v8bf y = *(const v8bf*)(p + 8);
  v16bf b;
#pragma unroll
  for (int e = 0; e < 8; ++e) { b[e] = x[e]; b[e + 8] = y[e]; }
  return b;
}

// ---------------------------------------------------------------------------
// Embedding gather + time-embedding broadcast add: h[t,:] = E[id[t],:] + te[b,:]
// ---------------------------------------------------------------------------
__global__ __launch_bounds__(256) void embed_kernel(const int* __restrict__ ids,
                                                    const float* __restrict__ emb,
                                                    const float* __restrict__ temb,
                                                    float* __restrict__ h) {
  const int t = blockIdx.x;
  const int b = t / cfg::S;
  const int id = ids[t];
  const float* er = emb + (size_t)id * cfg::H;
  const float* tr = temb + (size_t)b * cfg::H;
  float* hr = h + (size_t)t * cfg::H;
#pragma unroll
  for (int j = 0; j < cfg::H / 256; ++j) {
    const int i = threadIdx.x + j * 256;
    hr[i] = er[i] + tr[i];
  }
}

// ---------------------------------------------------------------------------
// RMSNorm: one block per token; optional bf16 and/or fp32 outputs
// ---------------------------------------------------------------------------
__global__ __launch_bounds__(256) void rmsnorm_kernel(const float* __restrict__ x,
                                                      const float* __restrict__ w,
                                                      bf16_t* __restrict__ outb,
                                                      float* __restrict__ outf) {
  __shared__ float red[256];
  const int t = blockIdx.x;
  const float* xr = x + (size_t)t * cfg::H;
  float ss = 0.0f;
#pragma unroll
  for (int j = 0; j < cfg::H / 256; ++j) {
    const float v = xr[threadIdx.x + j * 256];
    ss += v * v;
  }
  red[threadIdx.x] = ss;
  __syncthreads();
  for (int s = 128; s > 0; s >>= 1) {
    if (threadIdx.x < s) red[threadIdx.x] += red[threadIdx.x + s];
    __syncthreads();
  }
  const float rs = rsqrtf(red[0] * (1.0f / cfg::H) + cfg::EPS);
#pragma unroll
  for (int j = 0; j < cfg::H / 256; ++j) {
    const int i = threadIdx.x + j * 256;
    const float v = xr[i] * rs * w[i];
    if (outb) outb[(size_t)t * cfg::H + i] = (bf16_t)v;
    if (outf) outf[(size_t)t * cfg::H + i] = v;
  }
}

// ---------------------------------------------------------------------------
// RoPE in place on bf16 [T, nheads*128]
// ---------------------------------------------------------------------------
__global__ __launch_bounds__(256) void rope_kernel(bf16_t* __restrict__ x,
                                                   int nheads) {
  const int idx = blockIdx.x * blockDim.x + threadIdx.x;
  const int total = cfg::T * nheads * 64;
  if (idx >= total) return;
  const int i = idx & 63;
  const int hh = (idx >> 6) % nheads;
  const int t = idx / (64 * nheads);
  const int pos = t % cfg::S;
  // theta = pos * ROPE_BASE^(-2i/D) ; 2*ln(10000)/128 = 0.143911568
  const float theta = (float)pos * __expf(-0.143911568f * (float)i);
  const float c = __cosf(theta), s = __sinf(theta);
  bf16_t* p = x + (size_t)t * nheads * 128 + hh * 128;
  const float x1 = (float)p[i];
  const float x2 = (float)p[i + 64];
  p[i] = (bf16_t)(x1 * c - x2 * s);
  p[i + 64] = (bf16_t)(x2 * c + x1 * s);
}

// ---------------------------------------------------------------------------
// GEMM: C[M,N] = A(bf16,[M,K]) x W(fp32,[K,N])
// Block tile 64(M) x 128(N), K-step 32, 8 waves (2x4), wave tile 32x32.
// A tile staged via GLOBAL_LOAD_ASYNC_TO_LDS_B128 (pure bf16 copy).
// W tile converted fp32->bf16 in VGPRs and stored *transposed* [n][k] so
// B-fragment reads are two aligned ds_load_b128 instead of 16 scalar gathers.
// ---------------------------------------------------------------------------
enum { EP_STORE_BF16 = 0, EP_BIAS_SILU_BF16 = 1, EP_RESID_F32 = 2 };

template <int EPI>
__global__ __launch_bounds__(256) void gemm_bf16_kernel(
    const bf16_t* __restrict__ A, const float* __restrict__ W,
    const float* __restrict__ bias, float* __restrict__ resid,
    bf16_t* __restrict__ outb, int M, int N, int K) {
  constexpr int LDA = 48;   // 32 + pad (96B rows: 16B-aligned chunks)
  constexpr int LDBT = 48;  // transposed B tile: [128 n][48 k]
  __shared__ bf16_t As[64 * LDA];
  __shared__ bf16_t Bt[128 * LDBT];

  const int tid = threadIdx.x;
  const int lane = tid & 31;
  const int wave = tid >> 5;
  const int wm = wave >> 2;  // 0..1
  const int wn = wave & 3;   // 0..3
  const int m0 = blockIdx.y * 64;
  const int n0 = blockIdx.x * 128;

  v8f acc[2][2];
#pragma unroll
  for (int i = 0; i < 2; ++i)
#pragma unroll
    for (int j = 0; j < 2; ++j) acc[i][j] = vzero8();

  for (int k0 = 0; k0 < K; k0 += 32) {
    __syncthreads();
    {  // stage A tile (bf16, 64x32): one async b128 per thread
      const int r = tid >> 2;
      const int c = (tid & 3) * 8;
      async_copy_b128(&As[r * LDA + c], A + (size_t)(m0 + r) * K + k0 + c);
    }
    {  // stage B tile: fp32 -> bf16, transposed into [n][k]
      const int r = tid >> 3;         // k row 0..31
      const int c = (tid & 7) * 16;   // n base
      const float* src = W + (size_t)(k0 + r) * N + n0 + c;
#pragma unroll
      for (int j = 0; j < 16; ++j) Bt[(c + j) * LDBT + r] = (bf16_t)src[j];
    }
    if (k0 + 32 < K)  // gfx1250 global_prefetch for the next weight slice
      __builtin_prefetch(W + (size_t)(k0 + 32) * N + n0, 0, 0);
    wait_async();
    __syncthreads();

    const v16bf a0 = load_frag_a(As, LDA, wm * 32, 0, lane);
    const v16bf a1 = load_frag_a(As, LDA, wm * 32 + 16, 0, lane);
    const v16bf b0 = load_frag_b_nk(Bt, LDBT, wn * 32, 0, lane);
    const v16bf b1 = load_frag_b_nk(Bt, LDBT, wn * 32 + 16, 0, lane);
    acc[0][0] = wmma_bf16(a0, b0, acc[0][0]);
    acc[0][1] = wmma_bf16(a0, b1, acc[0][1]);
    acc[1][0] = wmma_bf16(a1, b0, acc[1][0]);
    acc[1][1] = wmma_bf16(a1, b1, acc[1][1]);
  }

  // Epilogue: C layout — vgpr r, lanes 0-15 -> row r, lanes 16-31 -> row r+8
  const int half8 = (lane >> 4) << 3;
#pragma unroll
  for (int mt = 0; mt < 2; ++mt)
#pragma unroll
    for (int nt = 0; nt < 2; ++nt)
#pragma unroll
      for (int r = 0; r < 8; ++r) {
        const int row = m0 + wm * 32 + mt * 16 + r + half8;
        const int col = n0 + wn * 32 + nt * 16 + (lane & 15);
        const float vacc = acc[mt][nt][r];
        if constexpr (EPI == EP_STORE_BF16) {
          outb[(size_t)row * N + col] = (bf16_t)vacc;
        } else if constexpr (EPI == EP_BIAS_SILU_BF16) {
          const float tv = vacc + bias[col];
          outb[(size_t)row * N + col] = (bf16_t)(tv / (1.0f + __expf(-tv)));
        } else {  // EP_RESID_F32
          const float tv = vacc + (bias ? bias[col] : 0.0f);
          resid[(size_t)row * N + col] += tv;
        }
      }
}

// ---------------------------------------------------------------------------
// Flash attention, causal + padding mask, GQA (4 q-heads per kv-head).
// Block: 128 threads = 4 waves; block tile = 64 queries; wave owns 16 queries.
// Q/K/V tiles staged via async global->LDS; Q*K^T and P*V via WMMA bf16;
// online softmax in fp32 wave-locally.
// ---------------------------------------------------------------------------
__global__ __launch_bounds__(128) void attn_kernel(
    const bf16_t* __restrict__ q, const bf16_t* __restrict__ k,
    const bf16_t* __restrict__ v, const float* __restrict__ amask,
    bf16_t* __restrict__ out) {
  constexpr int LDT = 136;  // 128 + pad
  constexpr int LDP = 72;   // 64 + pad
  __shared__ bf16_t Qs[64 * LDT];
  __shared__ bf16_t Ks[64 * LDT];
  __shared__ bf16_t Vs[64 * LDT];
  __shared__ bf16_t Ps[4 * 16 * LDP];

  const int tid = threadIdx.x;
  const int lane = tid & 31;
  const int wq = tid >> 5;  // wave id = which 16 queries
  const int qtile = blockIdx.x;
  const int head = blockIdx.y;
  const int batch = blockIdx.z;
  const int kvh = head >> 2;  // GROUPS = 4
  const int t0 = qtile * 64;
  const float scale = 0.08838834764831845f;  // 1/sqrt(128)
  const int half8 = (lane >> 4) << 3;

  {  // stage Q tile [64 x 128] via async copies
    const int r = tid >> 1;
    const int cb = (tid & 1) * 64;
    const bf16_t* src =
        q + (size_t)(batch * cfg::S + t0 + r) * (cfg::NH * cfg::D) + head * cfg::D + cb;
#pragma unroll
    for (int j = 0; j < 8; ++j)
      async_copy_b128(&Qs[r * LDT + cb + j * 8], src + j * 8);
  }
  wait_async();
  __syncthreads();

  float rowm[8], rowl[8];
  v8f O[8];
#pragma unroll
  for (int r = 0; r < 8; ++r) { rowm[r] = -1e30f; rowl[r] = 0.0f; }
#pragma unroll
  for (int nt = 0; nt < 8; ++nt) O[nt] = vzero8();

  for (int kb = 0; kb <= qtile; ++kb) {
    __syncthreads();
    {  // stage K,V tiles [64 x 128] via async copies
      const int r = tid >> 1;
      const int cb = (tid & 1) * 64;
      const size_t roff =
          (size_t)(batch * cfg::S + kb * 64 + r) * (cfg::NKV * cfg::D) + kvh * cfg::D + cb;
#pragma unroll
      for (int j = 0; j < 8; ++j) {
        async_copy_b128(&Ks[r * LDT + cb + j * 8], k + roff + j * 8);
        async_copy_b128(&Vs[r * LDT + cb + j * 8], v + roff + j * 8);
      }
    }
    wait_async();
    __syncthreads();

    // ---- scores S = Q*K^T (16 queries x 64 keys) ----
    v8f s[4];
#pragma unroll
    for (int f = 0; f < 4; ++f) s[f] = vzero8();
#pragma unroll
    for (int d0 = 0; d0 < 128; d0 += 32) {
      const v16bf aq = load_frag_a(Qs, LDT, wq * 16, d0, lane);
#pragma unroll
      for (int f = 0; f < 4; ++f) {
        const v16bf bk = load_frag_b_nk(Ks, LDT, f * 16, d0, lane);
        s[f] = wmma_bf16(aq, bk, s[f]);
      }
    }

    // ---- scale + masks + online softmax (wave-local) ----
    float amt[4];
#pragma unroll
    for (int f = 0; f < 4; ++f) {
      const int key = kb * 64 + f * 16 + (lane & 15);
      amt[f] = (1.0f - amask[batch * cfg::S + key]) * -10000.0f;
    }
#pragma unroll
    for (int r = 0; r < 8; ++r) {
      const int qrow = t0 + wq * 16 + r + half8;
      float rmax = -1e30f;
#pragma unroll
      for (int f = 0; f < 4; ++f) {
        const int key = kb * 64 + f * 16 + (lane & 15);
        float vs = s[f][r] * scale + amt[f];
        if (key > qrow) vs = -1e30f;
        s[f][r] = vs;
        rmax = fmaxf(rmax, vs);
      }
#pragma unroll
      for (int off = 1; off < 16; off <<= 1)
        rmax = fmaxf(rmax, __shfl_xor(rmax, off, 32));
      const float mnew = fmaxf(rowm[r], rmax);
      const float alpha = __expf(rowm[r] - mnew);
      float rsum = 0.0f;
#pragma unroll
      for (int f = 0; f < 4; ++f) {
        const float p = __expf(s[f][r] - mnew);
        s[f][r] = p;
        rsum += p;
      }
#pragma unroll
      for (int off = 1; off < 16; off <<= 1) rsum += __shfl_xor(rsum, off, 32);
      rowl[r] = rowl[r] * alpha + rsum;
      rowm[r] = mnew;
#pragma unroll
      for (int nt = 0; nt < 8; ++nt) O[nt][r] *= alpha;
    }

    // ---- P (fp32 C-layout) -> bf16 A-layout via per-wave LDS strip ----
    // same-wave LDS ops are in-order (DScnt in-order), no barrier needed
    bf16_t* Pw = Ps + wq * 16 * LDP;
#pragma unroll
    for (int r = 0; r < 8; ++r)
#pragma unroll
      for (int f = 0; f < 4; ++f)
        Pw[(r + half8) * LDP + f * 16 + (lane & 15)] = (bf16_t)s[f][r];

    // ---- O += P * V ----
#pragma unroll
    for (int ks = 0; ks < 2; ++ks) {
      const v16bf ap = load_frag_a(Pw, LDP, 0, ks * 32, lane);
#pragma unroll
      for (int nt = 0; nt < 8; ++nt) {
        const v16bf bv = load_frag_b_kn(Vs, LDT, nt * 16, ks * 32, lane);
        O[nt] = wmma_bf16(ap, bv, O[nt]);
      }
    }
  }

  // ---- normalize and store (bf16, [T, NH*D] with head-major cols) ----
#pragma unroll
  for (int r = 0; r < 8; ++r) {
    const float inv = 1.0f / rowl[r];
    const int trow = batch * cfg::S + t0 + wq * 16 + r + half8;
#pragma unroll
    for (int nt = 0; nt < 8; ++nt) {
      out[(size_t)trow * (cfg::NH * cfg::D) + head * cfg::D + nt * 16 + (lane & 15)] =
          (bf16_t)(O[nt][r] * inv);
    }
  }
}

// ---------------------------------------------------------------------------
// Host-side orchestration
// ---------------------------------------------------------------------------
extern "C" void kernel_launch(void* const* d_in, const int* in_sizes, int n_in,
                              void* d_out, int out_size, void* d_ws, size_t ws_size,
                              hipStream_t stream) {
  using namespace cfg;
  const int* input_ids  = (const int*)d_in[0];
  const float* amask    = (const float*)d_in[1];
  const float* time_emb = (const float*)d_in[2];
  const float* embed    = (const float*)d_in[3];
  const float* Wq = (const float*)d_in[4];
  const float* Wk = (const float*)d_in[5];
  const float* Wv = (const float*)d_in[6];
  const float* Wo = (const float*)d_in[7];
  const float* W1 = (const float*)d_in[8];
  const float* b1 = (const float*)d_in[9];
  const float* W2 = (const float*)d_in[10];
  const float* b2 = (const float*)d_in[11];
  const float* n1 = (const float*)d_in[12];
  const float* n2 = (const float*)d_in[13];
  const float* norm_w = (const float*)d_in[14];

  // workspace layout (bytes)
  constexpr size_t MiB = 1ull << 20;
  char* ws = (char*)d_ws;
  float*  h    = (float*)(ws);                 // [T,H]   fp32  32 MiB
  bf16_t* xn   = (bf16_t*)(ws + 32 * MiB);     // [T,H]   bf16  16 MiB
  bf16_t* qb   = (bf16_t*)(ws + 48 * MiB);     // [T,H]   bf16  16 MiB
  bf16_t* kbuf = (bf16_t*)(ws + 64 * MiB);     // [T,512] bf16   4 MiB
  bf16_t* vbuf = (bf16_t*)(ws + 68 * MiB);     // [T,512] bf16   4 MiB
  bf16_t* attn = (bf16_t*)(ws + 72 * MiB);     // [T,H]   bf16  16 MiB
  bf16_t* mlp  = (bf16_t*)(ws + 88 * MiB);     // [T,I]   bf16  64 MiB
  (void)ws_size; (void)in_sizes; (void)n_in; (void)out_size;

  const int KV = NKV * D;  // 512

  embed_kernel<<<T, 256, 0, stream>>>(input_ids, embed, time_emb, h);

  for (int l = 0; l < L; ++l) {
    const float* wq = Wq + (size_t)l * H * H;
    const float* wk = Wk + (size_t)l * H * KV;
    const float* wv = Wv + (size_t)l * H * KV;
    const float* wo = Wo + (size_t)l * H * H;
    const float* w1 = W1 + (size_t)l * H * I;
    const float* w2 = W2 + (size_t)l * I * H;
    const float* bb1 = b1 + (size_t)l * I;
    const float* bb2 = b2 + (size_t)l * H;

    rmsnorm_kernel<<<T, 256, 0, stream>>>(h, n1 + (size_t)l * H, xn, nullptr);

    gemm_bf16_kernel<EP_STORE_BF16><<<dim3(H / 128, T / 64), 256, 0, stream>>>(
        xn, wq, nullptr, nullptr, qb, T, H, H);
    gemm_bf16_kernel<EP_STORE_BF16><<<dim3(KV / 128, T / 64), 256, 0, stream>>>(
        xn, wk, nullptr, nullptr, kbuf, T, KV, H);
    gemm_bf16_kernel<EP_STORE_BF16><<<dim3(KV / 128, T / 64), 256, 0, stream>>>(
        xn, wv, nullptr, nullptr, vbuf, T, KV, H);

    rope_kernel<<<(T * NH * 64) / 256, 256, 0, stream>>>(qb, NH);
    rope_kernel<<<(T * NKV * 64) / 256, 256, 0, stream>>>(kbuf, NKV);

    attn_kernel<<<dim3(S / 64, NH, B), 128, 0, stream>>>(qb, kbuf, vbuf, amask, attn);

    gemm_bf16_kernel<EP_RESID_F32><<<dim3(H / 128, T / 64), 256, 0, stream>>>(
        attn, wo, nullptr, h, nullptr, T, H, H);

    rmsnorm_kernel<<<T, 256, 0, stream>>>(h, n2 + (size_t)l * H, xn, nullptr);

    gemm_bf16_kernel<EP_BIAS_SILU_BF16><<<dim3(I / 128, T / 64), 256, 0, stream>>>(
        xn, w1, bb1, nullptr, mlp, T, I, H);
    gemm_bf16_kernel<EP_RESID_F32><<<dim3(H / 128, T / 64), 256, 0, stream>>>(
        mlp, w2, bb2, h, nullptr, T, H, I);
  }

  rmsnorm_kernel<<<T, 256, 0, stream>>>(h, norm_w, nullptr, (float*)d_out);
}
